// GCNeXt_4861902979213
// MI455X (gfx1250) — matrix operand order repaired and checked by reading, not compiled, not run on hardware
//
#include <hip/hip_runtime.h>
#include <hip/hip_bf16.h>

typedef __attribute__((ext_vector_type(16))) __bf16 v16bf;
typedef __attribute__((ext_vector_type(8)))  __bf16 v8bf;
typedef __attribute__((ext_vector_type(4)))  __bf16 v4bf;
typedef __attribute__((ext_vector_type(8)))  float  v8f;
typedef __attribute__((ext_vector_type(4)))  float  v4f;
typedef unsigned int u32x4 __attribute__((ext_vector_type(4)));
typedef int          i32x8 __attribute__((ext_vector_type(8)));
typedef int          i32x4 __attribute__((ext_vector_type(4)));

constexpr int B_  = 8;
constexpr int C_  = 256;
constexpr int N_  = 4096;
constexpr int K_  = 3;
constexpr int G_  = 32;
constexpr int WID_ = 128;
constexpr int CO_ = 256;

// LDS panel: 128 rows, row stride 264 bf16 (528 B) -> rows shift 4 banks, so
// ds_load_b128 of 16 consecutive rows is bank-conflict free.
constexpr int PROW_ = 264;                       // elements per padded LDS row
constexpr int PANEL_BYTES_ = 128 * PROW_ * 2;    // 67584
constexpr int DTILE_BYTES_ = 128 * 132 * 4;      // 67584
constexpr int SMEM_K2_ = PANEL_BYTES_ + DTILE_BYTES_;

// ---------------- fragment loaders ----------------
// A fragment 16x32 bf16, contiguous k order used here (B same): lane<16 -> row M=lane, k [0..15]
//                             lane>=16 -> row M=lane-16, k [16..31]
// caller passes base = rowptr + kc + (hi?16:0); we read [0..15].
__device__ inline v16bf load_afrag_bf(const __bf16* p) {
  v8bf a0 = *(const v8bf*)(p);
  v8bf a1 = *(const v8bf*)(p + 8);
  return __builtin_shufflevector(a0, a1, 0,1,2,3,4,5,6,7,8,9,10,11,12,13,14,15);
}
__device__ inline v16bf load_afrag_f32(const float* p) {
  v4f f0 = *(const v4f*)(p);
  v4f f1 = *(const v4f*)(p + 4);
  v4f f2 = *(const v4f*)(p + 8);
  v4f f3 = *(const v4f*)(p + 12);
  v16bf a;
#pragma unroll
  for (int i = 0; i < 4; ++i) a[i]      = (__bf16)f0[i];
#pragma unroll
  for (int i = 0; i < 4; ++i) a[4 + i]  = (__bf16)f1[i];
#pragma unroll
  for (int i = 0; i < 4; ++i) a[8 + i]  = (__bf16)f2[i];
#pragma unroll
  for (int i = 0; i < 4; ++i) a[12 + i] = (__bf16)f3[i];
  return a;
}
__device__ inline v8f vzero8() { v8f z = {0.f,0.f,0.f,0.f,0.f,0.f,0.f,0.f}; return z; }

__device__ inline void topk_insert(float v, int j, float tv[3], int ti[3]) {
  bool g0 = (v > tv[0]) || (v == tv[0] && j < ti[0]);
  bool g1 = (v > tv[1]) || (v == tv[1] && j < ti[1]);
  bool g2 = (v > tv[2]) || (v == tv[2] && j < ti[2]);
  if (g0)      { tv[2]=tv[1]; ti[2]=ti[1]; tv[1]=tv[0]; ti[1]=ti[0]; tv[0]=v; ti[0]=j; }
  else if (g1) { tv[2]=tv[1]; ti[2]=ti[1]; tv[1]=v; ti[1]=j; }
  else if (g2) { tv[2]=v; ti[2]=j; }
}

// ---------------- K1: transpose (B,C,N) f32 -> (B,N,C) bf16 ----------------
__global__ __launch_bounds__(256) void k_transpose(const float* __restrict__ x,
                                                   __bf16* __restrict__ Xr) {
  __shared__ float tile[64][65];
  const int b  = blockIdx.z;
  const int c0 = blockIdx.y * 64;
  const int n0 = blockIdx.x * 64;
  const int tid = threadIdx.x;
#pragma unroll
  for (int p = 0; p < 16; ++p) {
    int e = tid + p * 256;
    int cl = e >> 6, nl = e & 63;
    tile[cl][nl] = x[((size_t)b * C_ + (c0 + cl)) * N_ + (n0 + nl)];
  }
  __syncthreads();
#pragma unroll
  for (int p = 0; p < 16; ++p) {
    int e = tid + p * 256;
    int nl = e >> 6, cl = e & 63;
    Xr[((size_t)b * N_ + (n0 + nl)) * C_ + (c0 + cl)] = (__bf16)tile[cl][nl];
  }
}

// ---------------- K1b: xx[b,n] = sum_c bf16(x)^2 ----------------
__global__ __launch_bounds__(256) void k_xx(const __bf16* __restrict__ Xr,
                                            float* __restrict__ xx) {
  const int t = blockIdx.x * 256 + threadIdx.x; // flat b*N + n
  const __bf16* p = Xr + (size_t)t * C_;
  float s = 0.f;
  for (int i = 0; i < C_; i += 16) {
    v16bf v = *(const v16bf*)(p + i);
#pragma unroll
    for (int q = 0; q < 16; ++q) { float f = (float)v[q]; s += f * f; }
  }
  xx[t] = s;
}

// ---------------- K2: fused Gram (WMMA bf16, LDS/TDM-staged B) + top-3 ----------------
__global__ __launch_bounds__(256) void k_gram_topk(const __bf16* __restrict__ Xr,
                                                   const float* __restrict__ xx,
                                                   int* __restrict__ idx) {
  extern __shared__ char smem[];
  __bf16* Bp = (__bf16*)smem;                     // padded 128 x PROW_ panel
  float*  D  = (float*)(smem + PANEL_BYTES_);     // 128 x 132 distance tile

  const int b    = blockIdx.y;
  const int m0   = blockIdx.x * 128;
  const int tid  = threadIdx.x;
  const int lane = tid & 31;
  const int wave = tid >> 5;
  const int l16  = lane & 15;
  const int hi   = lane >> 4;

  const __bf16* Xb  = Xr + (size_t)b * N_ * C_;
  const float*  xxb = xx + (size_t)b * N_;

  const int arow = m0 + wave * 16 + l16;
  const __bf16* Arow = Xb + (size_t)arow * C_ + (hi ? 16 : 0);
  const int mloc0 = wave * 16 + (hi ? 8 : 0);

  // Hoist the wave's entire A panel (16 rows x 256 k) into registers: 8 frags.
  v16bf afr[8];
#pragma unroll
  for (int kk = 0; kk < 8; ++kk) afr[kk] = load_afrag_bf(Arow + kk * 32);

  float xxm[8];
#pragma unroll
  for (int r = 0; r < 8; ++r) xxm[r] = xxb[m0 + mloc0 + r];

  float tv[3] = {-3.0e38f, -3.0e38f, -3.0e38f};
  int   ti[3] = {0, 0, 0};
  const int srow  = tid & 127;
  const int shalf = tid >> 7;

  for (int nb = 0; nb < N_ / 128; ++nb) {
    const int n0 = nb * 128;
    const __bf16* src = Xb + (size_t)n0 * C_;

    __syncthreads(); // previous iteration's panel reads + D scan complete

#if __has_builtin(__builtin_amdgcn_tensor_load_to_lds) && __has_builtin(__builtin_amdgcn_s_wait_tensorcnt)
    // Tensor Data Mover: one wave DMAs the 128x512B panel into LDS with
    // 16B padding every 128 dwords (row stride 528B in LDS).
    if (tid < 32) {
      unsigned long long ga = (unsigned long long)(uintptr_t)src;
      unsigned ldsa = (unsigned)(uintptr_t)Bp;    // generic->LDS: low 32 bits
      u32x4 g0;
      g0[0] = 1u;                                  // count=1, user descriptor
      g0[1] = ldsa;                                // lds_addr (bytes)
      g0[2] = (unsigned)(ga & 0xffffffffu);        // global_addr[31:0]
      g0[3] = (unsigned)((ga >> 32) & 0x01ffffffu) // global_addr[56:32]
              | (2u << 30);                        // type=2 ("image")
      i32x8 g1;
      g1[0] = (int)((2u << 16)                     // data_size = 4B
              | (1u << 20)                         // pad_enable
              | (6u << 22)                         // pad_interval: 128 dwords
              | (3u << 25));                       // pad_amount: 4 dwords (16B)
      g1[1] = (int)(128u << 16);                   // tensor_dim0 = 128 elems (lo16)
      g1[2] = (int)(4096u << 16);                  // dim0 hi=0 | tensor_dim1 lo16
      g1[3] = (int)(128u << 16);                   // dim1 hi=0 | tile_dim0 = 128
      g1[4] = 128;                                 // tile_dim1 = 128 rows
      g1[5] = 128;                                 // tensor_dim0_stride = 128 elems
      g1[6] = 0;
      g1[7] = 0;
      i32x4 z4 = {0, 0, 0, 0};
      i32x8 z8 = {0, 0, 0, 0, 0, 0, 0, 0};
      __builtin_amdgcn_tensor_load_to_lds(g0, g1, z4, z4, z8, 0);
      __builtin_amdgcn_s_wait_tensorcnt(0);
    }
#else
    // Fallback: cooperative load + ds_store staging (row pitch PROW_).
    for (int e = tid; e < 4096; e += 256) {
      int r = e >> 5, c = e & 31;                  // 32 x 16B chunks per row
      *(uint4*)((char*)Bp + r * (PROW_ * 2) + c * 16) =
          *(const uint4*)((const char*)src + r * 512 + c * 16);
    }
    if (nb + 1 < N_ / 128)
      __builtin_prefetch(src + (size_t)128 * C_ + (size_t)(tid >> 1) * C_ + (tid & 1) * 128, 0, 1);
#endif
    __syncthreads();

    v8f acc[8];
#pragma unroll
    for (int t = 0; t < 8; ++t) acc[t] = vzero8();

#pragma unroll
    for (int kk = 0; kk < 8; ++kk) {
      const int kc = kk * 32;
#pragma unroll
      for (int t = 0; t < 8; ++t) {
        const __bf16* bp = Bp + (t * 16 + l16) * PROW_ + kc + (hi ? 16 : 0);
        v16bf bb = *(const v16bf*)bp;
        acc[t] = __builtin_amdgcn_wmma_f32_16x16x32_bf16(
            false, afr[kk], false, bb, (short)0, acc[t], false, false);
      }
    }

#pragma unroll
    for (int t = 0; t < 8; ++t) {
      float xxn = xxb[n0 + t * 16 + l16];
#pragma unroll
      for (int r = 0; r < 8; ++r) {
        D[(mloc0 + r) * 132 + t * 16 + l16] = 2.0f * acc[t][r] - xxm[r] - xxn;
      }
    }
    __syncthreads();

    const float* rowp = &D[srow * 132 + shalf * 64];
    const int jb = n0 + shalf * 64;
    for (int c2 = 0; c2 < 64; ++c2) {
      topk_insert(rowp[c2], jb + c2, tv, ti);
    }
  }

  // merge the two half-row scanners
  __syncthreads();
  if (shalf == 1) {
    float* s = &D[srow * 8];
    s[0] = tv[0]; s[1] = tv[1]; s[2] = tv[2];
    ((int*)s)[3] = ti[0]; ((int*)s)[4] = ti[1]; ((int*)s)[5] = ti[2];
  }
  __syncthreads();
  if (shalf == 0) {
    const float* s = &D[srow * 8];
#pragma unroll
    for (int q = 0; q < 3; ++q) topk_insert(s[q], ((const int*)s)[3 + q], tv, ti);
    const size_t m = (size_t)b * N_ + (m0 + srow);
    idx[m * 3 + 0] = ti[0];
    idx[m * 3 + 1] = ti[1];
    idx[m * 3 + 2] = ti[2];
  }
}

// ---------------- K3: ya = w1a@x ; yb = w1b@x + b1 (WMMA) ----------------
__global__ __launch_bounds__(256) void k_w1(const __bf16* __restrict__ Xr,
                                            const float* __restrict__ w1,
                                            const float* __restrict__ b1,
                                            float* __restrict__ ya,
                                            float* __restrict__ yb) {
  const int b    = blockIdx.y;
  const int n0   = blockIdx.x * 128;
  const int tid  = threadIdx.x;
  const int lane = tid & 31;
  const int wave = tid >> 5;
  const int l16  = lane & 15;
  const int hi   = lane >> 4;

  const __bf16* Xb = Xr + (size_t)b * N_ * C_;
  const int orow  = wave * 16 + l16;
  const int obase = wave * 16 + (hi ? 8 : 0);

  for (int half = 0; half < 2; ++half) {
    v8f acc[8];
#pragma unroll
    for (int t = 0; t < 8; ++t) acc[t] = vzero8();

    for (int kc = 0; kc < C_; kc += 32) {
      const float* ap = w1 + (size_t)orow * (2 * C_) + half * C_ + kc + (hi ? 16 : 0);
      v16bf a = load_afrag_f32(ap);
#pragma unroll
      for (int t = 0; t < 8; ++t) {
        const __bf16* bp = Xb + (size_t)(n0 + t * 16 + l16) * C_ + kc + (hi ? 16 : 0);
        v16bf bb = *(const v16bf*)bp;
        acc[t] = __builtin_amdgcn_wmma_f32_16x16x32_bf16(
            false, a, false, bb, (short)0, acc[t], false, false);
      }
    }

    float* dst = (half == 0) ? ya : yb;
    v4f b1lo = *(const v4f*)(b1 + obase);
    v4f b1hi = *(const v4f*)(b1 + obase + 4);
#pragma unroll
    for (int t = 0; t < 8; ++t) {
      const int n = n0 + t * 16 + l16;
      float* d = dst + ((size_t)b * N_ + n) * WID_ + obase;
      v4f lo, hiv;
#pragma unroll
      for (int r = 0; r < 4; ++r) lo[r]  = acc[t][r]     + (half ? b1lo[r] : 0.f);
#pragma unroll
      for (int r = 0; r < 4; ++r) hiv[r] = acc[t][4 + r] + (half ? b1hi[r] : 0.f);
      *(v4f*)(d)     = lo;
      *(v4f*)(d + 4) = hiv;
    }
  }
}

// ---------------- K4: gather + relu + grouped conv (w2) + relu -> h2 bf16 ----------------
__global__ __launch_bounds__(256) void k_gather_gc(const int* __restrict__ idx,
                                                   const float* __restrict__ ya,
                                                   const float* __restrict__ yb,
                                                   const float* __restrict__ w2,
                                                   const float* __restrict__ b2,
                                                   __bf16* __restrict__ h2) {
  size_t t = (size_t)blockIdx.x * 256 + threadIdx.x; // ((b*N+n)*3 + k)*32 + g
  const int g = (int)(t & 31); t >>= 5;
  const int k = (int)(t % 3);  t /= 3;
  const size_t bn = t; // b*N + n
  const int j = idx[bn * 3 + k];
  const size_t bbase = (bn / N_) * (size_t)N_;
  const float* pa = ya + (bbase + j) * WID_ + g * 4;
  const float* pb = yb + bn * WID_ + g * 4;
  v4f va = *(const v4f*)pa;
  v4f vb = *(const v4f*)pb;
  float h[4];
#pragma unroll
  for (int i = 0; i < 4; ++i) h[i] = fmaxf(va[i] + vb[i], 0.f);
  const float* wg = w2 + g * 16;
  v4f b2v = *(const v4f*)(b2 + g * 4);
  v4bf res;
#pragma unroll
  for (int o = 0; o < 4; ++o) {
    float s = b2v[o];
#pragma unroll
    for (int i = 0; i < 4; ++i) s += wg[o * 4 + i] * h[i];
    res[o] = (__bf16)fmaxf(s, 0.f);
  }
  *(v4bf*)(h2 + (bn * 3 + k) * WID_ + g * 4) = res;
}

// ---------------- K5: h3 = relu(w3@h2 + b3), sout = max_k h3 (WMMA) ----------------
__global__ __launch_bounds__(256) void k_w3max(const __bf16* __restrict__ h2,
                                               const float* __restrict__ w3,
                                               const float* __restrict__ b3,
                                               float* __restrict__ sout) {
  const int b    = blockIdx.y;
  const int n0   = blockIdx.x * 64;
  const int tid  = threadIdx.x;
  const int lane = tid & 31;
  const int wave = tid >> 5;
  const int l16  = lane & 15;
  const int hi   = lane >> 4;

  float mx[2][4][8];
#pragma unroll
  for (int ch = 0; ch < 2; ++ch)
#pragma unroll
    for (int tn = 0; tn < 4; ++tn)
#pragma unroll
      for (int r = 0; r < 8; ++r) mx[ch][tn][r] = 0.f;

  float b3v[2][8];
#pragma unroll
  for (int ch = 0; ch < 2; ++ch) {
    const int cobase = wave * 16 + ch * 128 + (hi ? 8 : 0);
    v4f lo = *(const v4f*)(b3 + cobase);
    v4f hiv = *(const v4f*)(b3 + cobase + 4);
#pragma unroll
    for (int r = 0; r < 4; ++r) { b3v[ch][r] = lo[r]; b3v[ch][4 + r] = hiv[r]; }
  }

  for (int k = 0; k < 3; ++k) {
    v8f acc[2][4];
#pragma unroll
    for (int ch = 0; ch < 2; ++ch)
#pragma unroll
      for (int tn = 0; tn < 4; ++tn) acc[ch][tn] = vzero8();

    for (int oc = 0; oc < WID_; oc += 32) {
      v16bf a0 = load_afrag_f32(w3 + (size_t)(wave * 16 + l16) * WID_ + oc + (hi ? 16 : 0));
      v16bf a1 = load_afrag_f32(w3 + (size_t)(128 + wave * 16 + l16) * WID_ + oc + (hi ? 16 : 0));
#pragma unroll
      for (int tn = 0; tn < 4; ++tn) {
        const __bf16* bp = h2 + (((size_t)b * N_ + n0 + tn * 16 + l16) * 3 + k) * WID_
                              + oc + (hi ? 16 : 0);
        v16bf bb = *(const v16bf*)bp;
        acc[0][tn] = __builtin_amdgcn_wmma_f32_16x16x32_bf16(
            false, a0, false, bb, (short)0, acc[0][tn], false, false);
        acc[1][tn] = __builtin_amdgcn_wmma_f32_16x16x32_bf16(
            false, a1, false, bb, (short)0, acc[1][tn], false, false);
      }
    }
#pragma unroll
    for (int ch = 0; ch < 2; ++ch)
#pragma unroll
      for (int tn = 0; tn < 4; ++tn)
#pragma unroll
        for (int r = 0; r < 8; ++r) {
          float v = fmaxf(acc[ch][tn][r] + b3v[ch][r], 0.f);
          mx[ch][tn][r] = fmaxf(mx[ch][tn][r], v);
        }
  }

#pragma unroll
  for (int ch = 0; ch < 2; ++ch) {
    const int cobase = wave * 16 + ch * 128 + (hi ? 8 : 0);
#pragma unroll
    for (int tn = 0; tn < 4; ++tn) {
      const int n = n0 + tn * 16 + l16;
      float* d = sout + ((size_t)b * N_ + n) * CO_ + cobase;
      v4f lo, hiv;
#pragma unroll
      for (int r = 0; r < 4; ++r) { lo[r] = mx[ch][tn][r]; hiv[r] = mx[ch][tn][4 + r]; }
      *(v4f*)(d)     = lo;
      *(v4f*)(d + 4) = hiv;
    }
  }
}

// ---------------- K6: out = relu(wa@(x+sout) + ba) (WMMA) ----------------
__global__ __launch_bounds__(256) void k_final(const __bf16* __restrict__ Xr,
                                               const float* __restrict__ sout,
                                               const float* __restrict__ wa,
                                               const float* __restrict__ ba,
                                               float* __restrict__ out) {
  const int b    = blockIdx.y;
  const int n0   = blockIdx.x * 64;
  const int tid  = threadIdx.x;
  const int lane = tid & 31;
  const int wave = tid >> 5;
  const int l16  = lane & 15;
  const int hi   = lane >> 4;

  v8f acc[2][4];
#pragma unroll
  for (int ch = 0; ch < 2; ++ch)
#pragma unroll
    for (int tn = 0; tn < 4; ++tn) acc[ch][tn] = vzero8();

  for (int cc = 0; cc < C_; cc += 32) {
    v16bf a0 = load_afrag_f32(wa + (size_t)(wave * 16 + l16) * C_ + cc + (hi ? 16 : 0));
    v16bf a1 = load_afrag_f32(wa + (size_t)(128 + wave * 16 + l16) * C_ + cc + (hi ? 16 : 0));
#pragma unroll
    for (int tn = 0; tn < 4; ++tn) {
      const int n = n0 + tn * 16 + l16;
      const __bf16* xp = Xr + ((size_t)b * N_ + n) * C_ + cc + (hi ? 16 : 0);
      const float*  sp = sout + ((size_t)b * N_ + n) * CO_ + cc + (hi ? 16 : 0);
      v16bf xv = *(const v16bf*)xp;
      v16bf bb;
#pragma unroll
      for (int i = 0; i < 16; ++i) bb[i] = (__bf16)((float)xv[i] + sp[i]);
      acc[0][tn] = __builtin_amdgcn_wmma_f32_16x16x32_bf16(
          false, a0, false, bb, (short)0, acc[0][tn], false, false);
      acc[1][tn] = __builtin_amdgcn_wmma_f32_16x16x32_bf16(
          false, a1, false, bb, (short)0, acc[1][tn], false, false);
    }
  }

#pragma unroll
  for (int ch = 0; ch < 2; ++ch) {
    const int cobase = wave * 16 + ch * 128 + (hi ? 8 : 0);
    v4f balo = *(const v4f*)(ba + cobase);
    v4f bahi = *(const v4f*)(ba + cobase + 4);
#pragma unroll
    for (int tn = 0; tn < 4; ++tn) {
      const int n = n0 + tn * 16 + l16;
#pragma unroll
      for (int r = 0; r < 8; ++r) {
        const int co = cobase + r;
        float bias = (r < 4) ? balo[r] : bahi[r - 4];
        out[((size_t)b * CO_ + co) * N_ + n] = fmaxf(acc[ch][tn][r] + bias, 0.f);
      }
    }
  }
}

// ---------------- launch ----------------
extern "C" void kernel_launch(void* const* d_in, const int* in_sizes, int n_in,
                              void* d_out, int out_size, void* d_ws, size_t ws_size,
                              hipStream_t stream) {
  const float* x  = (const float*)d_in[0];
  const float* w1 = (const float*)d_in[1];
  const float* b1 = (const float*)d_in[2];
  const float* w2 = (const float*)d_in[3];
  const float* b2 = (const float*)d_in[4];
  const float* w3 = (const float*)d_in[5];
  const float* b3 = (const float*)d_in[6];
  const float* wa = (const float*)d_in[7];
  const float* ba = (const float*)d_in[8];
  float* out = (float*)d_out;

  char* ws = (char*)d_ws;
  size_t off = 0;
  __bf16* Xr = (__bf16*)(ws + off); off += (size_t)B_ * N_ * C_ * 2;          // 16 MB
  float*  xx = (float*)(ws + off);  off += (size_t)B_ * N_ * 4;               // 128 KB
  int*    idx = (int*)(ws + off);   off += (size_t)B_ * N_ * K_ * 4;          // 384 KB
  float*  ya = (float*)(ws + off);  off += (size_t)B_ * N_ * WID_ * 4;        // 16 MB
  float*  yb = (float*)(ws + off);  off += (size_t)B_ * N_ * WID_ * 4;        // 16 MB
  __bf16* h2 = (__bf16*)(ws + off); off += (size_t)B_ * N_ * K_ * WID_ * 2;   // 24 MB
  float*  so = (float*)(ws + off);  off += (size_t)B_ * N_ * CO_ * 4;         // 32 MB

  k_transpose<<<dim3(N_ / 64, C_ / 64, B_), 256, 0, stream>>>(x, Xr);
  k_xx<<<(B_ * N_) / 256, 256, 0, stream>>>(Xr, xx);
  k_gram_topk<<<dim3(N_ / 128, B_), 256, SMEM_K2_, stream>>>(Xr, xx, idx);
  k_w1<<<dim3(N_ / 128, B_), 256, 0, stream>>>(Xr, w1, b1, ya, yb);
  k_gather_gc<<<(B_ * N_ * K_ * G_) / 256, 256, 0, stream>>>(idx, ya, yb, w2, b2, h2);
  k_w3max<<<dim3(N_ / 64, B_), 256, 0, stream>>>(h2, w3, b3, so);
  k_final<<<dim3(N_ / 64, B_), 256, 0, stream>>>(Xr, so, wa, ba, out);
}
